// RouteNetModel_61048665145646
// MI455X (gfx1250) — matrix operand order repaired and checked by reading, not compiled, not run on hardware
//
#include <hip/hip_runtime.h>

#define NLINKS 20000
#define NPATHS 100000
#define PLEN   8
#define TITERS 8

typedef _Float16 h16;
typedef __attribute__((ext_vector_type(8)))  _Float16 v8h;
typedef __attribute__((ext_vector_type(16))) _Float16 v16h;
typedef __attribute__((ext_vector_type(8)))  float    v8f;

// ---------------------------------------------------------------- helpers ---

__device__ __forceinline__ v8f wmma_f16(v16h a, v16h b, v8f c) {
  // D = A(16x32 f16) * B(32x16 f16) + C(16x16 f32)
  return __builtin_amdgcn_wmma_f32_16x16x32_f16(false, a, false, b, (short)0, c,
                                                false, false);
}

__device__ __forceinline__ v8f splat8(float x) {
  v8f c;
#pragma unroll
  for (int i = 0; i < 8; ++i) c[i] = x;
  return c;
}

// Hardware transcendental path: v_rcp_f32 / v_exp_f32 (no IEEE div sequence).
__device__ __forceinline__ float fast_rcp(float x) {
  return __builtin_amdgcn_rcpf(x);
}
__device__ __forceinline__ float sigm(float x) {
  return fast_rcp(1.0f + __expf(-x));
}
__device__ __forceinline__ float tanh_fast(float x) {
  // exp->inf  => rcp->0   => +1 ; exp->0 => 1-2 = -1  (endpoints safe)
  return 1.0f - 2.0f * fast_rcp(__expf(2.0f * x) + 1.0f);
}

// Build a 16-bit A fragment (16x32, MxK) from an LDS tile with row stride
// `stride` halfs. Per ISA layout: lane holds row=lane&15; lanes>=16 use K+8.
// Halfs 0..7 = K hi+0..7, halfs 8..15 = K hi+16..23 -> two contiguous 16B loads.
__device__ __forceinline__ v16h a_frag_lds(const h16* base, int stride, int lane) {
  const int row = lane & 15;
  const int hi  = (lane >> 4) << 3;
  const h16* r = base + row * stride + hi;
  v8h lo = *(const v8h*)(r);
  v8h hh = *(const v8h*)(r + 16);
  v16h a;
#pragma unroll
  for (int i = 0; i < 8; ++i) { a[i] = lo[i]; a[i + 8] = hh[i]; }
  return a;
}

// GRU elementwise core on one 16-row tile held in C-layout registers.
// z/r gates: sigmoid(x@W + h@U + bi + bh) (biases pre-summed, chained C).
// cand: tanh(gi + r * gh) with gi/gh kept separate.
__device__ __forceinline__ void gru_tile(v16h ax, v16h ah,
                                         const v16h wf[6], const v16h uf[6],
                                         const float bsum[4], const float bi45[2],
                                         const float bh45[2], v8f& h0, v8f& h1,
                                         const float mk[8], bool use_mask) {
  v8f z0 = wmma_f16(ah, uf[0], wmma_f16(ax, wf[0], splat8(bsum[0])));
  v8f z1 = wmma_f16(ah, uf[1], wmma_f16(ax, wf[1], splat8(bsum[1])));
  v8f r0 = wmma_f16(ah, uf[2], wmma_f16(ax, wf[2], splat8(bsum[2])));
  v8f r1 = wmma_f16(ah, uf[3], wmma_f16(ax, wf[3], splat8(bsum[3])));
  v8f gi4 = wmma_f16(ax, wf[4], splat8(bi45[0]));
  v8f gh4 = wmma_f16(ah, uf[4], splat8(bh45[0]));
  v8f gi5 = wmma_f16(ax, wf[5], splat8(bi45[1]));
  v8f gh5 = wmma_f16(ah, uf[5], splat8(bh45[1]));
#pragma unroll
  for (int v = 0; v < 8; ++v) {
    float z0v = sigm(z0[v]), z1v = sigm(z1[v]);
    float r0v = sigm(r0[v]), r1v = sigm(r1[v]);
    float c0 = tanh_fast(gi4[v] + r0v * gh4[v]);
    float c1 = tanh_fast(gi5[v] + r1v * gh5[v]);
    float n0 = z0v * h0[v] + (1.0f - z0v) * c0;
    float n1 = z1v * h1[v] + (1.0f - z1v) * c1;
    if (use_mask) {
      n0 = (mk[v] != 0.0f) ? n0 : h0[v];
      n1 = (mk[v] != 0.0f) ? n1 : h1[v];
    }
    h0[v] = n0;
    h1[v] = n1;
  }
}

// ---------------------------------------------------------- setup kernels ---

// Permute a row-major [rows, ncols] f32 weight into WMMA B-fragment order:
// dst[((n*kchunks+kc)*32 + lane)*16 + i] = src[(kc*32 + i + koff(lane)) * ncols
//                                              + n*16 + (lane&15)]
__global__ void convert_frag_kernel(const float* __restrict__ src,
                                    h16* __restrict__ dst, int ncols, int ntiles,
                                    int kchunks) {
  int t = blockIdx.x * blockDim.x + threadIdx.x;
  int total = ntiles * kchunks * 32 * 16;
  if (t >= total) return;
  int i    = t & 15;
  int lane = (t >> 4) & 31;
  int nk   = t >> 9;
  int kc   = nk % kchunks;
  int n    = nk / kchunks;
  int koff = (lane >> 4) * 8 + (i >= 8 ? 8 : 0);
  int K    = kc * 32 + i + koff;
  int col  = n * 16 + (lane & 15);
  dst[t] = (h16)src[K * ncols + col];
}

__global__ void init_path_kernel(const float* __restrict__ traffic,
                                 const float* __restrict__ packets,
                                 const float* __restrict__ eqlam,
                                 const float* __restrict__ avgpkt,
                                 float* __restrict__ path_state) {
  int t = blockIdx.x * blockDim.x + threadIdx.x;
  if (t >= NPATHS * 32) return;
  int p = t >> 5, c = t & 31;
  float v = 0.0f;
  if (c == 0) v = traffic[p];
  else if (c == 1) v = packets[p];
  else if (c == 2) v = eqlam[p];
  else if (c == 3) v = avgpkt[p];
  path_state[t] = v;
}

__global__ void init_link_kernel(const float* __restrict__ cap,
                                 const float* __restrict__ qs,
                                 float* __restrict__ link_state) {
  int t = blockIdx.x * blockDim.x + threadIdx.x;
  if (t >= NLINKS * 32) return;
  int l = t >> 5, c = t & 31;
  float v = 0.0f;
  if (c == 0) v = cap[l];
  else if (c == 1) v = qs[l];
  link_state[t] = v;
}

__global__ void zero_kernel(float* __restrict__ p, int n) {
  int t = blockIdx.x * blockDim.x + threadIdx.x;
  if (t < n) p[t] = 0.0f;
}

// --------------------------------------------------------------- path GRU ---

__global__ __launch_bounds__(256) void path_gru_kernel(
    const float* __restrict__ link_state, float* __restrict__ path_state,
    const int* __restrict__ link_to_path, const h16* __restrict__ WpF,
    const h16* __restrict__ UpF, const float* __restrict__ bip,
    const float* __restrict__ bhp) {
  __shared__ __align__(16) h16 xb[8][16][32];
  __shared__ __align__(16) h16 hb[8][16][32];
  __shared__ float mb[8][16];

  const int lane = threadIdx.x & 31;
  const int wave = threadIdx.x >> 5;
  const int tile = blockIdx.x * 8 + wave;
  if (tile >= NPATHS / 16) return;
  const int p0   = tile * 16;
  const int nn   = lane & 15;
  const int m8   = (lane >> 4) << 3;
  const int half = lane >> 4;

  v16h wf[6], uf[6];
#pragma unroll
  for (int n = 0; n < 6; ++n) {
    wf[n] = *(const v16h*)(WpF + (n * 32 + lane) * 16);
    uf[n] = *(const v16h*)(UpF + (n * 32 + lane) * 16);
  }
  float bsum[4], bi45[2], bh45[2];
#pragma unroll
  for (int n = 0; n < 4; ++n) bsum[n] = bip[n * 16 + nn] + bhp[n * 16 + nn];
  bi45[0] = bip[64 + nn]; bi45[1] = bip[80 + nn];
  bh45[0] = bhp[64 + nn]; bh45[1] = bhp[80 + nn];

  // recurrent state in C-layout registers (cols 0..15 / 16..31)
  v8f h0, h1;
#pragma unroll
  for (int v = 0; v < 8; ++v) {
    h0[v] = path_state[(p0 + m8 + v) * 32 + nn];
    h1[v] = path_state[(p0 + m8 + v) * 32 + 16 + nn];
  }

  for (int s = 0; s < PLEN; ++s) {
    // gather one link state per row, stage f16 in LDS, track masking
    int li = link_to_path[(p0 + nn) * PLEN + s];
    const float4* src = (const float4*)(link_state + (long)li * 32) + half * 4;
    unsigned anyb = 0u;
#pragma unroll
    for (int q = 0; q < 4; ++q) {
      float4 f = src[q];
      anyb |= (__float_as_uint(f.x) | __float_as_uint(f.y) |
               __float_as_uint(f.z) | __float_as_uint(f.w)) & 0x7fffffffu;
      h16* d = &xb[wave][nn][half * 16 + q * 4];
      d[0] = (h16)f.x; d[1] = (h16)f.y; d[2] = (h16)f.z; d[3] = (h16)f.w;
    }
    unsigned oth = __shfl_xor(anyb, 16);
    mb[wave][nn] = ((anyb | oth) != 0u) ? 1.0f : 0.0f;

    // stage h (f16) for the h@Up operand
#pragma unroll
    for (int v = 0; v < 8; ++v) {
      hb[wave][m8 + v][nn]      = (h16)h0[v];
      hb[wave][m8 + v][16 + nn] = (h16)h1[v];
    }
    v16h ax = a_frag_lds(&xb[wave][0][0], 32, lane);
    v16h ah = a_frag_lds(&hb[wave][0][0], 32, lane);

    float mk[8];
#pragma unroll
    for (int v = 0; v < 8; ++v) mk[v] = mb[wave][m8 + v];

    gru_tile(ax, ah, wf, uf, bsum, bi45, bh45, h0, h1, mk, true);
  }

#pragma unroll
  for (int v = 0; v < 8; ++v) {
    path_state[(p0 + m8 + v) * 32 + nn]      = h0[v];
    path_state[(p0 + m8 + v) * 32 + 16 + nn] = h1[v];
  }
}

// --------------------------------------------------------- scatter to links --

__global__ void scatter_kernel(const float* __restrict__ path_state,
                               const int* __restrict__ link_to_path,
                               float* __restrict__ path_sum) {
  int t = blockIdx.x * blockDim.x + threadIdx.x;
  if (t >= NPATHS * PLEN * 8) return;
  int e = t >> 3;          // (path, seq) entry
  int q = t & 7;           // float4 chunk of the 32-wide state
  int l = link_to_path[e];
  int p = e >> 3;
  float4 v = ((const float4*)(path_state + (long)p * 32))[q];
  float* d = path_sum + (long)l * 32 + q * 4;
  unsafeAtomicAdd(d + 0, v.x);
  unsafeAtomicAdd(d + 1, v.y);
  unsafeAtomicAdd(d + 2, v.z);
  unsafeAtomicAdd(d + 3, v.w);
}

// --------------------------------------------------------------- link GRU ---

__global__ __launch_bounds__(256) void link_gru_kernel(
    const float* __restrict__ path_sum, float* __restrict__ link_state,
    const h16* __restrict__ WlF, const h16* __restrict__ UlF,
    const float* __restrict__ bil, const float* __restrict__ bhl) {
  __shared__ __align__(16) h16 xb[8][16][32];
  __shared__ __align__(16) h16 hb[8][16][32];

  const int lane = threadIdx.x & 31;
  const int wave = threadIdx.x >> 5;
  const int tile = blockIdx.x * 8 + wave;
  if (tile >= NLINKS / 16) return;
  const int l0   = tile * 16;
  const int nn   = lane & 15;
  const int m8   = (lane >> 4) << 3;
  const int half = lane >> 4;

  v16h wf[6], uf[6];
#pragma unroll
  for (int n = 0; n < 6; ++n) {
    wf[n] = *(const v16h*)(WlF + (n * 32 + lane) * 16);
    uf[n] = *(const v16h*)(UlF + (n * 32 + lane) * 16);
  }
  float bsum[4], bi45[2], bh45[2];
#pragma unroll
  for (int n = 0; n < 4; ++n) bsum[n] = bil[n * 16 + nn] + bhl[n * 16 + nn];
  bi45[0] = bil[64 + nn]; bi45[1] = bil[80 + nn];
  bh45[0] = bhl[64 + nn]; bh45[1] = bhl[80 + nn];

  // stage x = path_sum rows
  {
    const float4* src = (const float4*)(path_sum + (long)(l0 + nn) * 32) + half * 4;
#pragma unroll
    for (int q = 0; q < 4; ++q) {
      float4 f = src[q];
      h16* d = &xb[wave][nn][half * 16 + q * 4];
      d[0] = (h16)f.x; d[1] = (h16)f.y; d[2] = (h16)f.z; d[3] = (h16)f.w;
    }
  }
  v8f h0, h1;
#pragma unroll
  for (int v = 0; v < 8; ++v) {
    h0[v] = link_state[(l0 + m8 + v) * 32 + nn];
    h1[v] = link_state[(l0 + m8 + v) * 32 + 16 + nn];
  }
#pragma unroll
  for (int v = 0; v < 8; ++v) {
    hb[wave][m8 + v][nn]      = (h16)h0[v];
    hb[wave][m8 + v][16 + nn] = (h16)h1[v];
  }
  v16h ax = a_frag_lds(&xb[wave][0][0], 32, lane);
  v16h ah = a_frag_lds(&hb[wave][0][0], 32, lane);

  gru_tile(ax, ah, wf, uf, bsum, bi45, bh45, h0, h1, nullptr, false);

#pragma unroll
  for (int v = 0; v < 8; ++v) {
    link_state[(l0 + m8 + v) * 32 + nn]      = h0[v];
    link_state[(l0 + m8 + v) * 32 + 16 + nn] = h1[v];
  }
}

// ---------------------------------------------------------------- readout ---

__global__ __launch_bounds__(128) void readout_kernel(
    const float* __restrict__ link_state, const h16* __restrict__ W1F,
    const float* __restrict__ b1, const h16* __restrict__ W2F,
    const float* __restrict__ b2, const float* __restrict__ W3,
    const float* __restrict__ b3, float* __restrict__ out) {
  __shared__ __align__(16) h16 xb[4][16][32];
  __shared__ __align__(16) h16 r1b[4][16][256];

  const int lane = threadIdx.x & 31;
  const int wave = threadIdx.x >> 5;
  const int tile = blockIdx.x * 4 + wave;
  if (tile >= NLINKS / 16) return;
  const int l0   = tile * 16;
  const int nn   = lane & 15;
  const int m8   = (lane >> 4) << 3;
  const int half = lane >> 4;

  // stage link_state rows as f16
  {
    const float4* src = (const float4*)(link_state + (long)(l0 + nn) * 32) + half * 4;
#pragma unroll
    for (int q = 0; q < 4; ++q) {
      float4 f = src[q];
      h16* d = &xb[wave][nn][half * 16 + q * 4];
      d[0] = (h16)f.x; d[1] = (h16)f.y; d[2] = (h16)f.z; d[3] = (h16)f.w;
    }
  }
  v16h ax = a_frag_lds(&xb[wave][0][0], 32, lane);

  // layer 1: 32 -> 256, relu, stage f16
#pragma unroll
  for (int n = 0; n < 16; ++n) {
    v8f c = splat8(b1[n * 16 + nn]);
    v16h bfrag = *(const v16h*)(W1F + ((size_t)(n * 32 + lane)) * 16);
    c = wmma_f16(ax, bfrag, c);
#pragma unroll
    for (int v = 0; v < 8; ++v)
      r1b[wave][m8 + v][n * 16 + nn] = (h16)fmaxf(c[v], 0.0f);
  }

  // layer 2: 256 -> 256, relu, fold into out = r2 @ W3
  float acc[8];
#pragma unroll
  for (int v = 0; v < 8; ++v) acc[v] = 0.0f;

  for (int n = 0; n < 16; ++n) {
    v8f c = splat8(b2[n * 16 + nn]);
#pragma unroll
    for (int kc = 0; kc < 8; ++kc) {
      v16h a2 = a_frag_lds(&r1b[wave][0][0] + kc * 32, 256, lane);
      v16h bfrag = *(const v16h*)(W2F + ((size_t)((n * 8 + kc) * 32 + lane)) * 16);
      c = wmma_f16(a2, bfrag, c);
    }
    float w3v = W3[n * 16 + nn];
#pragma unroll
    for (int v = 0; v < 8; ++v) acc[v] += fmaxf(c[v], 0.0f) * w3v;
  }

  // reduce across the 16 columns held by each half-wave (lanes 0-15 / 16-31)
#pragma unroll
  for (int off = 1; off < 16; off <<= 1) {
#pragma unroll
    for (int v = 0; v < 8; ++v) acc[v] += __shfl_xor(acc[v], off);
  }
  if (nn == 0) {
#pragma unroll
    for (int v = 0; v < 8; ++v) out[l0 + m8 + v] = acc[v] + b3[0];
  }
}

// ------------------------------------------------------------------- host ---

extern "C" void kernel_launch(void* const* d_in, const int* in_sizes, int n_in,
                              void* d_out, int out_size, void* d_ws,
                              size_t ws_size, hipStream_t stream) {
  (void)in_sizes; (void)n_in; (void)out_size; (void)ws_size;

  const float* traffic = (const float*)d_in[0];
  const float* packets = (const float*)d_in[1];
  const float* eqlam   = (const float*)d_in[2];
  const float* avgpkt  = (const float*)d_in[3];
  const float* cap     = (const float*)d_in[4];
  const float* qs      = (const float*)d_in[5];
  const int*   l2p     = (const int*)d_in[6];
  // d_in[7..10] (path_ids, sequence_path, path_to_link, sequence_links) are
  // derivable: path_ids[e]=e/8, path_to_link==path_ids, sequence_links==l2p.
  const float* Wp = (const float*)d_in[13];
  const float* Up = (const float*)d_in[14];
  const float* bip = (const float*)d_in[15];
  const float* bhp = (const float*)d_in[16];
  const float* Wl = (const float*)d_in[17];
  const float* Ul = (const float*)d_in[18];
  const float* bil = (const float*)d_in[19];
  const float* bhl = (const float*)d_in[20];
  const float* W1 = (const float*)d_in[21];
  const float* b1 = (const float*)d_in[22];
  const float* W2 = (const float*)d_in[23];
  const float* b2 = (const float*)d_in[24];
  const float* W3 = (const float*)d_in[25];
  const float* b3 = (const float*)d_in[26];

  char* w = (char*)d_ws;
  auto take = [&](size_t bytes) {
    void* p = (void*)w;
    w += (bytes + 255) & ~(size_t)255;
    return p;
  };
  float* path_state = (float*)take((size_t)NPATHS * 32 * 4);
  float* link_state = (float*)take((size_t)NLINKS * 32 * 4);
  float* path_sum   = (float*)take((size_t)NLINKS * 32 * 4);
  h16* WpF = (h16*)take(6 * 32 * 16 * 2);
  h16* UpF = (h16*)take(6 * 32 * 16 * 2);
  h16* WlF = (h16*)take(6 * 32 * 16 * 2);
  h16* UlF = (h16*)take(6 * 32 * 16 * 2);
  h16* W1F = (h16*)take(16 * 32 * 16 * 2);
  h16* W2F = (h16*)take(16 * 8 * 32 * 16 * 2);

  // weight permutation into B-fragment order
  convert_frag_kernel<<<(6 * 512 + 255) / 256, 256, 0, stream>>>(Wp, WpF, 96, 6, 1);
  convert_frag_kernel<<<(6 * 512 + 255) / 256, 256, 0, stream>>>(Up, UpF, 96, 6, 1);
  convert_frag_kernel<<<(6 * 512 + 255) / 256, 256, 0, stream>>>(Wl, WlF, 96, 6, 1);
  convert_frag_kernel<<<(6 * 512 + 255) / 256, 256, 0, stream>>>(Ul, UlF, 96, 6, 1);
  convert_frag_kernel<<<(16 * 512 + 255) / 256, 256, 0, stream>>>(W1, W1F, 256, 16, 1);
  convert_frag_kernel<<<(16 * 8 * 512 + 255) / 256, 256, 0, stream>>>(W2, W2F, 256, 16, 8);

  init_path_kernel<<<(NPATHS * 32 + 255) / 256, 256, 0, stream>>>(
      traffic, packets, eqlam, avgpkt, path_state);
  init_link_kernel<<<(NLINKS * 32 + 255) / 256, 256, 0, stream>>>(cap, qs, link_state);

  const int pathTiles = NPATHS / 16;  // 6250
  const int linkTiles = NLINKS / 16;  // 1250

  for (int t = 0; t < TITERS; ++t) {
    path_gru_kernel<<<(pathTiles + 7) / 8, 256, 0, stream>>>(
        link_state, path_state, l2p, WpF, UpF, bip, bhp);
    zero_kernel<<<(NLINKS * 32 + 255) / 256, 256, 0, stream>>>(path_sum, NLINKS * 32);
    scatter_kernel<<<(NPATHS * PLEN * 8 + 255) / 256, 256, 0, stream>>>(
        path_state, l2p, path_sum);
    link_gru_kernel<<<(linkTiles + 7) / 8, 256, 0, stream>>>(
        path_sum, link_state, WlF, UlF, bil, bhl);
  }

  readout_kernel<<<(linkTiles + 3) / 4, 128, 0, stream>>>(
      link_state, W1F, b1, W2F, b2, W3, b3, (float*)d_out);
}
